// _BiLSTMCRFNetwork_67044439491144
// MI455X (gfx1250) — compile-verified, matrix operand order checked by reading, not run on hardware
//
#include <hip/hip_runtime.h>
#include <stdint.h>

// ---------------- CDNA5 WMMA types ----------------
typedef __attribute__((ext_vector_type(16))) __bf16 v16bf;
typedef __attribute__((ext_vector_type(8)))  float  v8f;

#define B_   32
#define T_   512
#define E_   256
#define H_   512
#define K_   17
#define WGS  8      // workgroups (hidden-slices) per direction in LSTM
#define SLICE 64    // hidden units per slice
#define NP   256    // 4*SLICE gate columns per slice

// ---------------- workspace layout (bytes) ----------------
#define OFF_SYNC  ((size_t)0)                         // 2*512 ints       = 4 KB
#define OFF_HBUF  ((size_t)4096)                      // 2dir*2par*32*512 bf16 = 128 KB
#define OFF_WIH   (OFF_HBUF + (size_t)131072)         // packed W_ih bf16 = 2 MB
#define OFF_WHH   (OFF_WIH  + (size_t)2097152)        // packed W_hh bf16 = 4 MB
#define OFF_XG    (OFF_WHH  + (size_t)4194304)        // gate preacts bf16 = 128 MB
#define OFF_ENC   (OFF_XG   + (size_t)134217728)      // enc bf16 [32,512,1024] = 32 MB
#define OFF_EMIS  (OFF_ENC  + (size_t)33554432)       // emissions f32 [32,512,17]

union Frag { uint4 q[2]; v16bf v; __bf16 e[16]; };
union Acc  { v8f v; float f[8]; };

// ---------------- CDNA5 async global->LDS copy (ASYNCcnt-tracked) ----------------
#if __has_builtin(__builtin_amdgcn_global_load_async_to_lds_b128)
#define ASYNC_LDS 1
typedef int v4i __attribute__((__vector_size__(16)));
typedef __attribute__((address_space(1))) v4i* as1_v4i;
typedef __attribute__((address_space(3))) v4i* as3_v4i;
#endif

__device__ __forceinline__ void ld_lds_b128(void* l, const void* g) {
#ifdef ASYNC_LDS
    __builtin_amdgcn_global_load_async_to_lds_b128((as1_v4i)(g), (as3_v4i)(l), 0, 0);
#else
    *(uint4*)l = *(const uint4*)g;
#endif
}

__device__ __forceinline__ void wait_lds_copies() {
#ifdef ASYNC_LDS
#if __has_builtin(__builtin_amdgcn_s_wait_asynccnt)
    __builtin_amdgcn_s_wait_asynccnt(0);
#else
    asm volatile("s_wait_asynccnt 0x0" ::: "memory");
#endif
#endif
}

// K offset inside a 16x32 bf16 A-tile (and mirrored B-tile) for packed
// element p (0..15) of a lane; half = lane>>4.
__device__ __forceinline__ int kofs(int p, int half) {
    int v = p >> 1, pair = p & 1;
    int base = (v < 4) ? (v * 2) : (16 + (v - 4) * 2);
    return base + half * 8 + pair;
}

__device__ __forceinline__ float sigf(float x) { return 1.f / (1.f + __expf(-x)); }

// ---------------- zero sync counters + h ping-pong buffers ----------------
__global__ void kZero(uint32_t* ws32) {
    int i = blockIdx.x * blockDim.x + threadIdx.x;
    if (i < 1024 + 32768) ws32[i] = 0u;   // sync (4KB) + hbuf (128KB), contiguous
}

// ---------------- pack W_ih (both dirs) into WMMA-B lane layout ----------------
// dst index = dir*524288 + kt*65536 + ntg*512 + lane*16 + p
__global__ void kPackIH(const float* __restrict__ wf, const float* __restrict__ wb,
                        __bf16* __restrict__ dst) {
    int idx = blockIdx.x * blockDim.x + threadIdx.x;
    if (idx >= 1048576) return;
    int p    = idx & 15;
    int lane = (idx >> 4) & 31;
    int ntg  = (idx >> 9) & 127;
    int kt   = (idx >> 16) & 7;
    int dir  = idx >> 19;
    int k = kt * 32 + kofs(p, lane >> 4);
    int n = ntg * 16 + (lane & 15);
    const float* w = dir ? wb : wf;
    dst[idx] = (__bf16)w[(size_t)n * E_ + k];
}

// ---------------- pack W_hh per (dir, slice) into WMMA-B lane layout ----------------
// dst index = dir*1048576 + wg*131072 + kt*8192 + nt*512 + lane*16 + p
__global__ void kPackHH(const float* __restrict__ wf, const float* __restrict__ wb,
                        __bf16* __restrict__ dst) {
    int idx = blockIdx.x * blockDim.x + threadIdx.x;
    if (idx >= 2097152) return;
    int p    = idx & 15;
    int lane = (idx >> 4) & 31;
    int nt   = (idx >> 9) & 15;
    int kt   = (idx >> 13) & 15;
    int wg   = (idx >> 17) & 7;
    int dir  = idx >> 20;
    int k  = kt * 32 + kofs(p, lane >> 4);
    int np = nt * 16 + (lane & 15);      // 0..255 within slice: [i|f|g|o] x 64
    int gate = np >> 6, jj = np & 63;
    int r = gate * H_ + wg * SLICE + jj; // original W_hh row
    const float* w = dir ? wb : wf;
    dst[idx] = (__bf16)w[(size_t)r * H_ + k];
}

// ---------------- embedding gather + input GEMM (xg = x @ W_ih^T + b) ----------------
// grid: 2 dirs x 512 mBlocks x 8 nBlocks. block = 256 thr = 8 waves.
// Wave computes a 16x64 C-block (1 M-tile x 4 N-tiles), k-loop of 8.
__global__ void __launch_bounds__(256) kInputGemm(
        const int* __restrict__ ids, const int* __restrict__ lengths,
        const float* __restrict__ emb, const float* __restrict__ bf,
        const float* __restrict__ bb, const __bf16* __restrict__ wihp,
        __bf16* __restrict__ xg) {
    int blk = blockIdx.x;
    int dir = blk >> 12;
    int rr  = blk & 4095;
    int mB  = rr >> 3, nB = rr & 7;
    int tid = threadIdx.x, wv = tid >> 5, lane = tid & 31, half = lane >> 4;
    int mt = wv >> 2, nts = (wv & 3) * 4;
    int mBase = mB * 32 + mt * 16;

    // A-row metadata for this lane (rows 0-15 of the tile)
    int mrow = mBase + (lane & 15);
    int b = mrow >> 9, t = mrow & 511;
    int tt = t;
    if (dir) { int len = lengths[b]; tt = (t < len) ? (len - 1 - t) : t; }
    const float* erow = emb + (size_t)ids[b * T_ + tt] * E_;

    Acc acc[4];
#pragma unroll
    for (int i = 0; i < 4; i++)
#pragma unroll
        for (int r = 0; r < 8; r++) acc[i].f[r] = 0.f;

#pragma unroll
    for (int kt = 0; kt < 8; kt++) {
        int kA = kt * 32 + half * 8;
        float4 f0 = *(const float4*)(erow + kA);
        float4 f1 = *(const float4*)(erow + kA + 4);
        float4 f2 = *(const float4*)(erow + kA + 16);
        float4 f3 = *(const float4*)(erow + kA + 20);
        Frag a;
        a.e[0]=(__bf16)f0.x; a.e[1]=(__bf16)f0.y; a.e[2]=(__bf16)f0.z; a.e[3]=(__bf16)f0.w;
        a.e[4]=(__bf16)f1.x; a.e[5]=(__bf16)f1.y; a.e[6]=(__bf16)f1.z; a.e[7]=(__bf16)f1.w;
        a.e[8]=(__bf16)f2.x; a.e[9]=(__bf16)f2.y; a.e[10]=(__bf16)f2.z; a.e[11]=(__bf16)f2.w;
        a.e[12]=(__bf16)f3.x; a.e[13]=(__bf16)f3.y; a.e[14]=(__bf16)f3.z; a.e[15]=(__bf16)f3.w;
#pragma unroll
        for (int i = 0; i < 4; i++) {
            int ntg = nB * 16 + nts + i;
            const __bf16* bp = wihp + (((size_t)(dir * 8 + kt) * 128 + ntg) * 32 + lane) * 16;
            Frag bfr;
            bfr.q[0] = *(const uint4*)bp;
            bfr.q[1] = *(const uint4*)(bp + 8);
            acc[i].v = __builtin_amdgcn_wmma_f32_16x16x32_bf16(
                false, a.v, false, bfr.v, (short)0, acc[i].v, false, false);
        }
    }

    // store: add bias, remap column -> (slice, np), write bf16 gate preacts
    const float* bias = dir ? bb : bf;
#pragma unroll
    for (int i = 0; i < 4; i++) {
        int ntg = nB * 16 + nts + i;
        int ng  = ntg * 16 + (lane & 15);       // global gate column 0..2047
        float bv = bias[ng];
        int gate = ng >> 9, j = ng & 511;
        int wg = j >> 6, np = gate * 64 + (j & 63);
#pragma unroll
        for (int r = 0; r < 8; r++) {
            int mr = mBase + r + 8 * half;
            int b2 = mr >> 9, t2 = mr & 511;
            size_t di = ((((size_t)(dir * 8 + wg) * T_ + t2) * B_) + b2) * NP + np;
            xg[di] = (__bf16)(acc[i].f[r] + bv);
        }
    }
}

// ---------------- recurrent BiLSTM scan ----------------
// 16 blocks: dir = blk>>3, slice wg = blk&7. W_hh slice (256KB) LDS-resident.
// Per step: 32x256 = h(t-1)[32,512] @ slice^T via 512 v_wmma, then gate math,
// h-slice published through L2, per-step grid barrier via agent atomics.
// LDS staging uses CDNA5 async global->LDS copies when available.
__global__ void __launch_bounds__(256) kLstm(
        const __bf16* __restrict__ xg, const __bf16* __restrict__ whhp,
        __bf16* __restrict__ hbuf, __bf16* __restrict__ enc,
        const int* __restrict__ lengths, int* __restrict__ sync) {
    int dir = blockIdx.x >> 3, wg = blockIdx.x & 7;
    int tid = threadIdx.x, wv = tid >> 5, lane = tid & 31, half = lane >> 4;
    int mt = wv >> 2, nts = (wv & 3) * 4;

    extern __shared__ char smem[];
    __bf16* sW = (__bf16*)smem;               // 131072 bf16 = 256 KB (weights, persistent)
    __bf16* sH = (__bf16*)(smem + 262144);    // 32x512 bf16 = 32 KB (h staging)
    float*  sG = (float*)(smem + 262144);     // overlay: 32x256 f32 gate exchange

    // one-time async fill of the LDS weight slice
    {
        const uint4* src = (const uint4*)whhp + (size_t)(dir * 8 + wg) * 16384;
        uint4* dst = (uint4*)sW;
        for (int i = tid; i < 16384; i += 256) ld_lds_b128(dst + i, src + i);
        wait_lds_copies();
    }

    float cst[8];
#pragma unroll
    for (int e = 0; e < 8; e++) cst[e] = 0.f;

    for (int t = 0; t < T_; t++) {
        // wait until all 8 slices of this direction published h(t-1)
        if (t > 0 && tid == 0) {
            while (__hip_atomic_load(&sync[dir * T_ + t - 1], __ATOMIC_ACQUIRE,
                                     __HIP_MEMORY_SCOPE_AGENT) < WGS)
                __builtin_amdgcn_s_sleep(2);
        }
        __syncthreads();

        // stage h(t-1) (ping-pong parity) into LDS via async copies
        {
            const uint4* hs = (const uint4*)(hbuf + (size_t)(dir * 2 + ((t + 1) & 1)) * 16384);
            uint4* hd = (uint4*)sH;
            for (int i = tid; i < 2048; i += 256) ld_lds_b128(hd + i, hs + i);
            wait_lds_copies();
        }
        const __bf16* xrow = xg + ((size_t)(dir * 8 + wg) * T_ + t) * B_ * NP;
        if (t + 1 < T_) {
            const char* pf = (const char*)(xrow + (size_t)B_ * NP);
            __builtin_prefetch(pf + tid * 64, 0, 0);   // global_prefetch_b8: next step's gates
        }
        __syncthreads();

        // init accumulators from precomputed input-gate preactivations
        Acc acc[4];
#pragma unroll
        for (int i = 0; i < 4; i++) {
            int np = (nts + i) * 16 + (lane & 15);
#pragma unroll
            for (int r = 0; r < 8; r++) {
                int brow = mt * 16 + r + 8 * half;
                acc[i].f[r] = (float)xrow[brow * NP + np];
            }
        }

        // G += h(t-1) @ W_hh_slice^T : 16 k-tiles x 4 n-tiles of v_wmma
        int m = mt * 16 + (lane & 15);
        for (int kt = 0; kt < 16; kt++) {
            int kA = kt * 32 + half * 8;
            Frag a;
            a.q[0] = *(const uint4*)(sH + m * H_ + kA);
            a.q[1] = *(const uint4*)(sH + m * H_ + kA + 16);
#pragma unroll
            for (int i = 0; i < 4; i++) {
                const __bf16* bp = sW + ((size_t)(kt * 16 + nts + i) * 32 + lane) * 16;
                Frag bfr;
                bfr.q[0] = *(const uint4*)bp;
                bfr.q[1] = *(const uint4*)(bp + 8);
                acc[i].v = __builtin_amdgcn_wmma_f32_16x16x32_bf16(
                    false, a.v, false, bfr.v, (short)0, acc[i].v, false, false);
            }
        }
        __syncthreads();   // done reading sH -> reuse region as f32 gate buffer

        // exchange gates through LDS (i/f/g/o of one hidden unit live in 4 tiles)
#pragma unroll
        for (int i = 0; i < 4; i++) {
            int np = (nts + i) * 16 + (lane & 15);
#pragma unroll
            for (int r = 0; r < 8; r++) {
                int brow = mt * 16 + r + 8 * half;
                sG[brow * NP + np] = acc[i].f[r];
            }
        }
        __syncthreads();

        // elementwise LSTM cell for this slice's 32x64 (b, j) pairs
        int wpar = dir * 2 + (t & 1);
#pragma unroll
        for (int e = 0; e < 8; e++) {
            int p = tid * 8 + e;
            int b = p >> 6, jl = p & 63;
            float iv = sG[b * NP + jl];
            float fv = sG[b * NP + 64 + jl];
            float gv = sG[b * NP + 128 + jl];
            float ov = sG[b * NP + 192 + jl];
            float c = sigf(fv) * cst[e] + sigf(iv) * tanhf(gv);
            cst[e] = c;
            float h = sigf(ov) * tanhf(c);
            __bf16 hb = (__bf16)h;
            int jg = wg * SLICE + jl;
            hbuf[(size_t)wpar * 16384 + b * H_ + jg] = hb;
            int pos = t;
            if (dir) { int len = lengths[b]; pos = (t < len) ? (len - 1 - t) : t; }
            enc[((size_t)b * T_ + pos) * 1024 + dir * H_ + jg] = hb;
        }
        __threadfence();
        __syncthreads();
        if (tid == 0)
            __hip_atomic_fetch_add(&sync[dir * T_ + t], 1, __ATOMIC_RELEASE,
                                   __HIP_MEMORY_SCOPE_AGENT);
    }
}

// ---------------- emissions: enc @ W_cls^T + b_cls (K=17) ----------------
// one wave per (b,t); lane L owns enc[L*32 .. L*32+32)
__global__ void __launch_bounds__(256) kEmis(
        const __bf16* __restrict__ enc, const float* __restrict__ wcls,
        const float* __restrict__ bcls, float* __restrict__ emis) {
    int wv = threadIdx.x >> 5, lane = threadIdx.x & 31;
    int gw = blockIdx.x * 8 + wv;                    // 0..16383 = b*512+t
    const __bf16* er = enc + (size_t)gw * 1024 + lane * 32;
    union { uint4 q[4]; __bf16 e[32]; } ev;
    ev.q[0] = ((const uint4*)er)[0];
    ev.q[1] = ((const uint4*)er)[1];
    ev.q[2] = ((const uint4*)er)[2];
    ev.q[3] = ((const uint4*)er)[3];
    for (int k = 0; k < K_; k++) {
        const float4* wr = (const float4*)(wcls + (size_t)k * 1024 + lane * 32);
        float s = 0.f;
#pragma unroll
        for (int i = 0; i < 8; i++) {
            float4 w4 = wr[i];
            s += (float)ev.e[i * 4 + 0] * w4.x + (float)ev.e[i * 4 + 1] * w4.y +
                 (float)ev.e[i * 4 + 2] * w4.z + (float)ev.e[i * 4 + 3] * w4.w;
        }
        for (int off = 16; off >= 1; off >>= 1) s += __shfl_xor(s, off, 32);
        if (lane == 0) emis[(size_t)gw * K_ + k] = s + bcls[k];
    }
}

// ---------------- CRF: numerator + forward algorithm + mean NLL ----------------
// one block, wave w handles batch b=w; lane j<17 carries alpha[j]
__global__ void __launch_bounds__(1024) kCrf(
        const float* __restrict__ emis, const int* __restrict__ tags,
        const int* __restrict__ lengths, const float* __restrict__ startt,
        const float* __restrict__ trans, const float* __restrict__ endt,
        float* __restrict__ out) {
    __shared__ float sT[K_ * K_];
    __shared__ float sL[32];
    int tid = threadIdx.x;
    if (tid < K_ * K_) sT[tid] = trans[tid];
    __syncthreads();
    int w = tid >> 5, lane = tid & 31;
    int b = w;
    int len = lengths[b];
    const float* Eb = emis + (size_t)b * T_ * K_;

    float alpha = -1e30f;
    if (lane < K_) alpha = startt[lane] + Eb[lane];
    int col = (lane < K_) ? lane : 0;

    for (int t = 1; t < T_; t++) {
        float e = (lane < K_) ? Eb[t * K_ + lane] : 0.f;
        float mx = -1e30f, s = 0.f;
        for (int i = 0; i < K_; i++) {
            float ai = __shfl(alpha, i, 32);
            float v = ai + sT[i * K_ + col];
            if (v > mx) { s = s * __expf(mx - v) + 1.f; mx = v; }
            else         s += __expf(v - mx);
        }
        float nxt = mx + __logf(s) + e;
        if (t < len && lane < K_) alpha = nxt;
    }

    float val = (lane < K_) ? (alpha + endt[lane]) : -1e30f;
    float m2 = val;
    for (int off = 16; off >= 1; off >>= 1) m2 = fmaxf(m2, __shfl_xor(m2, off, 32));
    float se = (lane < K_) ? __expf(val - m2) : 0.f;
    for (int off = 16; off >= 1; off >>= 1) se += __shfl_xor(se, off, 32);
    float logZ = m2 + __logf(se);

    const int* tg = tags + b * T_;
    float num = 0.f;
    for (int t = lane; t < T_; t += 32) {
        if (t < len) {
            int g = tg[t];
            num += Eb[t * K_ + g];
            if (t >= 1) num += sT[tg[t - 1] * K_ + g];
        }
    }
    for (int off = 16; off >= 1; off >>= 1) num += __shfl_xor(num, off, 32);
    if (lane == 0) {
        num += startt[tg[0]] + endt[tg[len - 1]];
        sL[w] = logZ - num;
    }
    __syncthreads();
    if (tid == 0) {
        float s = 0.f;
        for (int i = 0; i < 32; i++) s += sL[i];
        out[0] = s / 32.f;   // mean(logZ - numer) == -mean(numer - logZ)
    }
}

// ---------------- host ----------------
extern "C" void kernel_launch(void* const* d_in, const int* in_sizes, int n_in,
                              void* d_out, int out_size, void* d_ws, size_t ws_size,
                              hipStream_t stream) {
    const int*   ids     = (const int*)d_in[0];
    const int*   lengths = (const int*)d_in[1];
    const int*   tags    = (const int*)d_in[2];
    // d_in[3] = mask : recomputed from lengths, unused
    const float* emb     = (const float*)d_in[4];
    const float* wihf    = (const float*)d_in[5];
    const float* whhf    = (const float*)d_in[6];
    const float* bfv     = (const float*)d_in[7];
    const float* wihb    = (const float*)d_in[8];
    const float* whhb    = (const float*)d_in[9];
    const float* bbv     = (const float*)d_in[10];
    const float* wcls    = (const float*)d_in[11];
    const float* bcls    = (const float*)d_in[12];
    const float* stt     = (const float*)d_in[13];
    const float* trn     = (const float*)d_in[14];
    const float* ett     = (const float*)d_in[15];

    char* ws = (char*)d_ws;
    int*    syncp = (int*)(ws + OFF_SYNC);
    __bf16* hbuf  = (__bf16*)(ws + OFF_HBUF);
    __bf16* wihp  = (__bf16*)(ws + OFF_WIH);
    __bf16* whhp  = (__bf16*)(ws + OFF_WHH);
    __bf16* xg    = (__bf16*)(ws + OFF_XG);
    __bf16* enc   = (__bf16*)(ws + OFF_ENC);
    float*  emis  = (float*)(ws + OFF_EMIS);

    kZero<<<132, 256, 0, stream>>>((uint32_t*)ws);
    kPackIH<<<4096, 256, 0, stream>>>(wihf, wihb, wihp);
    kPackHH<<<8192, 256, 0, stream>>>(whhf, whhb, whhp);
    kInputGemm<<<8192, 256, 0, stream>>>(ids, lengths, emb, bfv, bbv, wihp, xg);

    (void)hipFuncSetAttribute((const void*)kLstm,
                              hipFuncAttributeMaxDynamicSharedMemorySize, 294912);
    kLstm<<<16, 256, 294912, stream>>>(xg, whhp, hbuf, enc, lengths, syncp);

    kEmis<<<2048, 256, 0, stream>>>(enc, wcls, bcls, emis);
    kCrf<<<1, 1024, 0, stream>>>(emis, tags, lengths, stt, trn, ett, (float*)d_out);
}
